// BinConv2d_2173253452167
// MI455X (gfx1250) — matrix operand order repaired and checked by reading, not compiled, not run on hardware
//
#include <hip/hip_runtime.h>
#include <hip/hip_bf16.h>

typedef __attribute__((ext_vector_type(4)))  _Float16 v4h;
typedef __attribute__((ext_vector_type(8)))  _Float16 v8h;
typedef __attribute__((ext_vector_type(16))) _Float16 v16h;
typedef __attribute__((ext_vector_type(8)))  float    v8f;
typedef __attribute__((ext_vector_type(4)))  float    v4f;
typedef __attribute__((ext_vector_type(4)))  int      v4i;

typedef __attribute__((address_space(1))) v4i* gptr_v4i;   // global int4*
typedef __attribute__((address_space(3))) v4i* lptr_v4i;   // LDS int4*

#define EPSV   1e-4f
#define C_IN   64
#define C_OUT  64
#define HW     112
#define NB     32
#define KDIM   576              // 9 taps * 64 channels
#define AW     114              // width + 2 halo
#define NROWS  4                // staged input rows: h0-1 .. h0+2 (2 output rows)
#define A_HALVES (NROWS*AW*C_IN)  // 29184 halves = 58368 B
#define B_HALVES (C_OUT*KDIM)     // 36864 halves = 73728 B
#define NTHREADS 224

#if __has_builtin(__builtin_amdgcn_global_load_async_to_lds_b128) && \
    __has_builtin(__builtin_amdgcn_s_wait_asynccnt)
#define USE_ASYNC_LDS 1
#endif

__device__ __forceinline__ _Float16 sgnh(float t) {
  return (t > 0.f) ? (_Float16)1.f : ((t < 0.f) ? (_Float16)-1.f : (_Float16)0.f);
}

// ---------------- Kernel A: per-channel batch stats -> scale/shift ----------------
__global__ __launch_bounds__(256)
void bn_stats_kernel(const float* __restrict__ x,
                     const float* __restrict__ gamma,
                     const float* __restrict__ beta,
                     float* __restrict__ ws) {
  const int c   = blockIdx.x;
  const int tid = threadIdx.x;
  float sum = 0.f, sq = 0.f;
  for (int n = 0; n < NB; ++n) {
    const v4f* p = (const v4f*)(x + ((size_t)n * C_IN + c) * (HW * HW));
    for (int i = tid; i < (HW * HW) / 4; i += 256) {
      v4f v = p[i];
      sum += v[0] + v[1] + v[2] + v[3];
      sq  += v[0]*v[0] + v[1]*v[1] + v[2]*v[2] + v[3]*v[3];
    }
  }
  __shared__ float ssum[256], ssq[256];
  ssum[tid] = sum; ssq[tid] = sq;
  __syncthreads();
  for (int s = 128; s > 0; s >>= 1) {
    if (tid < s) { ssum[tid] += ssum[tid + s]; ssq[tid] += ssq[tid + s]; }
    __syncthreads();
  }
  if (tid == 0) {
    const float m    = (float)(NB * HW * HW);
    const float mean = ssum[0] / m;
    const float var  = ssq[0] / m - mean * mean;   // biased
    const float sc   = gamma[c] * rsqrtf(var + EPSV);
    ws[c]        = sc;
    ws[C_IN + c] = beta[c] - mean * sc;
  }
}

// ---------------- Kernel W: one-shot f32 -> f16 weight repack: Wh[co][p*64+c] ----------------
__global__ __launch_bounds__(256)
void wprep_kernel(const float* __restrict__ W, _Float16* __restrict__ Wh) {
  const int idx = blockIdx.x * 256 + threadIdx.x;   // 144*256 == 36864 exactly
  const int n = idx / KDIM;
  const int k = idx - n * KDIM;
  const int p = k >> 6;
  const int c = k & 63;
  Wh[idx] = (_Float16)W[((size_t)n * C_IN + c) * 9 + p];
}

// ---------------- Kernel B: fused sign-binarize + implicit-GEMM conv (WMMA f16) ----------------
// One block per (n, 2 output rows). 7 waves; wave wv owns w in [16wv,16wv+16) x 2 h x 64 co.
__global__ __launch_bounds__(NTHREADS)
void binconv_wmma_kernel(const float* __restrict__ x,
                         const _Float16* __restrict__ Wh,
                         const float* __restrict__ bias,
                         const float* __restrict__ ws,
                         float* __restrict__ out) {
  extern __shared__ _Float16 smem[];
  _Float16* As = smem;             // [4][114][64]  (r, w+1, c) binarized input
  _Float16* Bs = smem + A_HALVES;  // [64][576]     (co, k=p*64+c) f16 weights

  const int nb  = blockIdx.x / (HW / 2);
  const int h0  = (blockIdx.x % (HW / 2)) * 2;
  const int tid = threadIdx.x;
  const float* scale = ws;
  const float* shift = ws + C_IN;

  // ---- (1) kick off async weight copy global(f16, L2-resident) -> LDS ----
#ifdef USE_ASYNC_LDS
  for (int idx = tid; idx < B_HALVES / 8; idx += NTHREADS) {
    __builtin_amdgcn_global_load_async_to_lds_b128(
        (gptr_v4i)(size_t)(Wh + idx * 8),
        (lptr_v4i)(unsigned)(size_t)(Bs + idx * 8),
        0, 0);
  }
#else
  for (int idx = tid; idx < B_HALVES / 8; idx += NTHREADS)
    ((v8h*)Bs)[idx] = ((const v8h*)Wh)[idx];
#endif

  // ---- (2) zero As (covers halo + OOB rows), b128 stores ----
  for (int idx = tid; idx < A_HALVES / 8; idx += NTHREADS)
    ((v8h*)As)[idx] = (v8h)0.0f;
  __syncthreads();

  // ---- (3) interior fill: 4 rows x 16 c-quads x 28 w-quads = 1792 chunks ----
  for (int idx = tid; idx < NROWS * 16 * 28; idx += NTHREADS) {
    const int r   = idx / (16 * 28);
    const int rem = idx - r * (16 * 28);
    const int cq  = rem / 28;
    const int wq  = rem - cq * 28;
    const int c   = cq * 4;
    const int win = wq * 4;
    const int hin = h0 + r - 1;
    if ((unsigned)hin < (unsigned)HW) {
      v4f xv[4]; float sc[4], sh[4];
#pragma unroll
      for (int k = 0; k < 4; ++k) {
        xv[k] = *(const v4f*)&x[(((size_t)nb * C_IN + c + k) * HW + hin) * HW + win];
        sc[k] = scale[c + k]; sh[k] = shift[c + k];
      }
#pragma unroll
      for (int wj = 0; wj < 4; ++wj) {
        v4h st;
#pragma unroll
        for (int k = 0; k < 4; ++k) st[k] = sgnh(xv[k][wj] * sc[k] + sh[k]);
        *(v4h*)&As[(r * AW + win + 1 + wj) * C_IN + c] = st;   // ds_store_b64
      }
    }
  }
#ifdef USE_ASYNC_LDS
  __builtin_amdgcn_s_wait_asynccnt(0);
#endif
  __syncthreads();

  // ---- (4) WMMA mainloop: K = p*64+c, 18 steps of 32; 8 accumulators (2h x 4 co-tiles) ----
  const int lane = tid & 31;
  const int wv   = tid >> 5;          // 0..6
  const int w0   = wv * 16;
  const int half = lane >> 4;
  const int mrow = lane & 15;         // A row (w within segment)
  const int col  = lane & 15;         // B/C column (co within tile)

  v8f acc[8] = {};                    // [hh*4 + t]

#pragma unroll
  for (int p = 0; p < 9; ++p) {
    const int kh = p / 3, kw = p % 3;
#pragma unroll
    for (int cc = 0; cc < 2; ++cc) {
      const int c0 = cc * 32;
      const int krow = p * 64 + c0 + half * 16;
      const v16h b0 = *(const v16h*)&Bs[(col     ) * KDIM + krow];
      const v16h b1 = *(const v16h*)&Bs[(col + 16) * KDIM + krow];
      const v16h b2 = *(const v16h*)&Bs[(col + 32) * KDIM + krow];
      const v16h b3 = *(const v16h*)&Bs[(col + 48) * KDIM + krow];
#pragma unroll
      for (int hh = 0; hh < 2; ++hh) {
        const int arow = ((kh + hh) * AW + (w0 + mrow + kw)) * C_IN;
        v8h g0 = *(const v8h*)&As[arow + c0 + half * 16];
        v8h g1 = *(const v8h*)&As[arow + c0 + half * 16 + 8];
        v16h a;
#pragma unroll
        for (int e = 0; e < 8; ++e) { a[e] = g0[e]; a[8 + e] = g1[e]; }
        acc[hh*4+0] = __builtin_amdgcn_wmma_f32_16x16x32_f16(false, a, false, b0, (short)0, acc[hh*4+0], false, false);
        acc[hh*4+1] = __builtin_amdgcn_wmma_f32_16x16x32_f16(false, a, false, b1, (short)0, acc[hh*4+1], false, false);
        acc[hh*4+2] = __builtin_amdgcn_wmma_f32_16x16x32_f16(false, a, false, b2, (short)0, acc[hh*4+2], false, false);
        acc[hh*4+3] = __builtin_amdgcn_wmma_f32_16x16x32_f16(false, a, false, b3, (short)0, acc[hh*4+3], false, false);
      }
    }
  }

  // ---- (5) epilogue: +bias, ReLU, two b128 stores per tile ----
  const int wout = w0 + half * 8;     // C layout: M = (lane>>4)*8 + vgpr
#pragma unroll
  for (int hh = 0; hh < 2; ++hh) {
#pragma unroll
    for (int t = 0; t < 4; ++t) {
      const int co = t * 16 + col;
      const float bv = bias[co];
      float* op = out + (((size_t)nb * C_OUT + co) * HW + (h0 + hh)) * HW + wout;
      v4f lo, hi;
#pragma unroll
      for (int j = 0; j < 4; ++j) { lo[j] = fmaxf(acc[hh*4+t][j]     + bv, 0.f); }
#pragma unroll
      for (int j = 0; j < 4; ++j) { hi[j] = fmaxf(acc[hh*4+t][4 + j] + bv, 0.f); }
      *(v4f*)op       = lo;
      *(v4f*)(op + 4) = hi;
    }
  }
}

extern "C" void kernel_launch(void* const* d_in, const int* in_sizes, int n_in,
                              void* d_out, int out_size, void* d_ws, size_t ws_size,
                              hipStream_t stream) {
  const float* x     = (const float*)d_in[0];
  const float* gamma = (const float*)d_in[1];
  const float* beta  = (const float*)d_in[2];
  const float* W     = (const float*)d_in[3];
  const float* b     = (const float*)d_in[4];
  float* out = (float*)d_out;
  float* ws  = (float*)d_ws;                 // [0..63] scale, [64..127] shift
  _Float16* Wh = (_Float16*)(ws + 128);      // 36864 f16 repacked weights

  bn_stats_kernel<<<C_IN, 256, 0, stream>>>(x, gamma, beta, ws);
  wprep_kernel<<<B_HALVES / 256, 256, 0, stream>>>(W, Wh);

  const size_t shmem = (size_t)(A_HALVES + B_HALVES) * 2;   // 132,096 B < 160 KB
  binconv_wmma_kernel<<<NB * (HW / 2), NTHREADS, shmem, stream>>>(x, Wh, b, ws, out);
}
